// MalwareGNN_48610439856177
// MI455X (gfx1250) — matrix operand order repaired and checked by reading, not compiled, not run on hardware
//
#include <hip/hip_runtime.h>
#include <stdint.h>

#define N_NODES  100000
#define N_EDGES  320000
#define FDIM     256
#define N_GRAPHS 256
#define N_CLASSES 32
#define N_CENT   3
#define EPSV     1e-5f

typedef __attribute__((ext_vector_type(16))) __bf16     v16bf;
typedef __attribute__((ext_vector_type(8)))  float      v8f;
typedef __attribute__((ext_vector_type(8)))  uint32_t   v8u;

__device__ __forceinline__ unsigned short f2bf(float f) {
    uint32_t u = __builtin_bit_cast(uint32_t, f);
    uint32_t r = u + 0x7fffu + ((u >> 16) & 1u);   // round-to-nearest-even
    return (unsigned short)(r >> 16);
}

// ---------------- degree / dinv ----------------
__global__ void deg_init(float* dinv) {
    int i = blockIdx.x * blockDim.x + threadIdx.x;
    if (i < N_NODES) dinv[i] = 1.0f;               // self-loop
}
__global__ void deg_edges(const int* __restrict__ dst, float* dinv) {
    int e = blockIdx.x * blockDim.x + threadIdx.x;
    if (e < N_EDGES) atomicAdd(&dinv[dst[e]], 1.0f);
}
__global__ void deg_rsqrt(float* dinv) {
    int i = blockIdx.x * blockDim.x + threadIdx.x;
    if (i < N_NODES) dinv[i] = rsqrtf(dinv[i]);
}

// ---------------- pack W into WMMA B-fragment order ----------------
// packed layout: tile t = n_tile*8 + k_step (16 n_tiles x 8 k_steps);
// within tile: lane l (0..31) holds 16 contiguous bf16 = elems e 0..15
//   k = k_step*32 + (e&7) + (e>=8 ? 16 : 0) + (l>=16 ? 8 : 0),  n = n_tile*16 + (l&15)
__global__ void pack_w(const float* __restrict__ W, unsigned short* __restrict__ wp) {
    int idx = blockIdx.x * blockDim.x + threadIdx.x;      // 65536
    if (idx >= FDIM * FDIM) return;
    int tile = idx >> 9;
    int lane = (idx >> 4) & 31;
    int e    = idx & 15;
    int n_tile = tile >> 3;
    int k_step = tile & 7;
    int k = k_step * 32 + (e & 7) + ((e >= 8) ? 16 : 0) + ((lane >= 16) ? 8 : 0);
    int n = n_tile * 16 + (lane & 15);
    wp[idx] = f2bf(W[k * FDIM + n]);
}

// ---------------- GEMM: H[M,256] = X[M,256] @ W  (bf16 WMMA, f32 acc) ----------------
__global__ void __launch_bounds__(256) gemm_bf16(const float* __restrict__ X,
                                                 const unsigned short* __restrict__ WP,
                                                 float* __restrict__ H) {
    // raw f32 A panel (16 rows x 256 cols, contiguous in X) DMA'd by the async engine
    __shared__ __align__(16) float          fA[16 * FDIM];      // 16 KB
    // A panel repacked in WMMA A-fragment order: [k_step][lane][e] (bf16)
    __shared__ __align__(32) unsigned short lA[8 * 32 * 16];    //  8 KB
    const int tid  = threadIdx.x;
    const int row0 = blockIdx.x * 16;

    {   // ---- async DMA: 16 KB contiguous panel -> LDS, no VGPR data round-trip ----
        uint64_t gaddr = (uint64_t)(uintptr_t)(X + (size_t)row0 * FDIM) + (uint64_t)tid * 16;
        uint32_t laddr = (uint32_t)(uintptr_t)(&fA[0]) + (uint32_t)tid * 16;
        asm volatile("global_load_async_to_lds_b128 %0, %1, off"
                     :: "v"(laddr), "v"(gaddr) : "memory");
        asm volatile("global_load_async_to_lds_b128 %0, %1, off offset:4096"
                     :: "v"(laddr), "v"(gaddr) : "memory");
        asm volatile("global_load_async_to_lds_b128 %0, %1, off offset:8192"
                     :: "v"(laddr), "v"(gaddr) : "memory");
        asm volatile("global_load_async_to_lds_b128 %0, %1, off offset:12288"
                     :: "v"(laddr), "v"(gaddr) : "memory");
        asm volatile("s_wait_asynccnt 0x0" ::: "memory");
    }
    __syncthreads();

    {   // ---- repack to bf16 fragments: thread (k_step = tid>>5, lt = tid&31) ----
        int k_step = tid >> 5;
        int lt     = tid & 31;
        int m      = lt & 15;
        int off8   = (lt >= 16) ? 8 : 0;
        int kb     = k_step * 32;
        const float* xr = &fA[m * FDIM];
        float4 a0 = *(const float4*)(xr + kb + off8);
        float4 a1 = *(const float4*)(xr + kb + off8 + 4);
        float4 a2 = *(const float4*)(xr + kb + 16 + off8);
        float4 a3 = *(const float4*)(xr + kb + 16 + off8 + 4);
        unsigned short* d = &lA[(k_step * 32 + lt) * 16];
        d[0] = f2bf(a0.x); d[1] = f2bf(a0.y); d[2]  = f2bf(a0.z); d[3]  = f2bf(a0.w);
        d[4] = f2bf(a1.x); d[5] = f2bf(a1.y); d[6]  = f2bf(a1.z); d[7]  = f2bf(a1.w);
        d[8] = f2bf(a2.x); d[9] = f2bf(a2.y); d[10] = f2bf(a2.z); d[11] = f2bf(a2.w);
        d[12]= f2bf(a3.x); d[13]= f2bf(a3.y); d[14] = f2bf(a3.z); d[15] = f2bf(a3.w);
    }
    __syncthreads();

    const int wave = tid >> 5;       // 8 waves -> cols [wave*32, wave*32+32)
    const int lane = tid & 31;
    v8f c0 = {}; v8f c1 = {};
    const int t0 = (wave * 2) * 8;       // packed tile base, n_tile = wave*2
    const int t1 = (wave * 2 + 1) * 8;

    #pragma unroll
    for (int ks = 0; ks < 8; ++ks) {
        v16bf a  = __builtin_bit_cast(v16bf, *(const v8u*)&lA[(ks * 32 + lane) * 16]);
        v16bf b0 = __builtin_bit_cast(v16bf, *(const v8u*)&WP[((t0 + ks) * 32 + lane) * 16]);
        v16bf b1 = __builtin_bit_cast(v16bf, *(const v8u*)&WP[((t1 + ks) * 32 + lane) * 16]);
        if (ks < 7) {
            __builtin_prefetch(&WP[((t0 + ks + 1) * 32 + lane) * 16], 0, 3);
            __builtin_prefetch(&WP[((t1 + ks + 1) * 32 + lane) * 16], 0, 3);
        }
        c0 = __builtin_amdgcn_wmma_f32_16x16x32_bf16(false, a, false, b0, (short)0, c0, false, false);
        c1 = __builtin_amdgcn_wmma_f32_16x16x32_bf16(false, a, false, b1, (short)0, c1, false, false);
    }

    // C layout: VGPR r -> M = r + (lane>=16 ? 8 : 0), N = lane&15
    const int n0 = wave * 32 + (lane & 15);
    const int mo = (lane >= 16) ? 8 : 0;
    #pragma unroll
    for (int r = 0; r < 8; ++r) {
        size_t ro = (size_t)(row0 + r + mo) * FDIM;
        H[ro + n0]      = c0[r];
        H[ro + n0 + 16] = c1[r];
    }
}

// ---------------- aggregation ----------------
// O[i] = dinv[i]^2 * H[i]   (self-loop term), float4 granularity
__global__ void self_loop(const float* __restrict__ H, const float* __restrict__ dinv,
                          float* __restrict__ O) {
    size_t idx = (size_t)blockIdx.x * blockDim.x + threadIdx.x;   // N_NODES*64 float4s
    if (idx >= (size_t)N_NODES * (FDIM / 4)) return;
    int node = (int)(idx >> 6);
    float w = dinv[node]; w = w * w;
    float4 h = ((const float4*)H)[idx];
    float4 r; r.x = w * h.x; r.y = w * h.y; r.z = w * h.z; r.w = w * h.w;
    ((float4*)O)[idx] = r;
}

// one wave per edge; lane covers 8 features
__global__ void edge_scatter(const float* __restrict__ H, float* __restrict__ O,
                             const int* __restrict__ src, const int* __restrict__ dst,
                             const float* __restrict__ dinv) {
    int w = (blockIdx.x * blockDim.x + threadIdx.x) >> 5;
    int lane = threadIdx.x & 31;
    if (w >= N_EDGES) return;
    int s = src[w], d = dst[w];
    float norm = dinv[s] * dinv[d];
    const float4* hs = (const float4*)(H + (size_t)s * FDIM) + lane * 2;
    float4 v0 = hs[0], v1 = hs[1];
    float* o = O + (size_t)d * FDIM + lane * 8;
    atomicAdd(o + 0, norm * v0.x); atomicAdd(o + 1, norm * v0.y);
    atomicAdd(o + 2, norm * v0.z); atomicAdd(o + 3, norm * v0.w);
    atomicAdd(o + 4, norm * v1.x); atomicAdd(o + 5, norm * v1.y);
    atomicAdd(o + 6, norm * v1.z); atomicAdd(o + 7, norm * v1.w);
}

// in-place: x = relu?(((x + bias) - m) * rsqrt(v+eps) * g + beta)
__global__ void bias_bn(float* __restrict__ X, const float* __restrict__ bias,
                        const float* __restrict__ g, const float* __restrict__ beta,
                        const float* __restrict__ m, const float* __restrict__ v, int relu) {
    size_t idx = (size_t)blockIdx.x * blockDim.x + threadIdx.x;
    if (idx >= (size_t)N_NODES * (FDIM / 4)) return;
    int c4 = (int)(idx & 63) * 4;
    float4 x  = ((float4*)X)[idx];
    float4 bb = *(const float4*)(bias + c4);
    float4 gg = *(const float4*)(g + c4);
    float4 be = *(const float4*)(beta + c4);
    float4 mm = *(const float4*)(m + c4);
    float4 vv = *(const float4*)(v + c4);
    float4 r;
    r.x = (x.x + bb.x - mm.x) * rsqrtf(vv.x + EPSV) * gg.x + be.x;
    r.y = (x.y + bb.y - mm.y) * rsqrtf(vv.y + EPSV) * gg.y + be.y;
    r.z = (x.z + bb.z - mm.z) * rsqrtf(vv.z + EPSV) * gg.z + be.z;
    r.w = (x.w + bb.w - mm.w) * rsqrtf(vv.w + EPSV) * gg.w + be.w;
    if (relu) {
        r.x = fmaxf(r.x, 0.f); r.y = fmaxf(r.y, 0.f);
        r.z = fmaxf(r.z, 0.f); r.w = fmaxf(r.w, 0.f);
    }
    ((float4*)X)[idx] = r;
}

// ---------------- pooling ----------------
__global__ void pool_zero(float* sums, float* cnts) {
    int i = blockIdx.x * blockDim.x + threadIdx.x;
    if (i < N_GRAPHS * FDIM) sums[i] = 0.f;
    if (i < N_GRAPHS) cnts[i] = 0.f;
}
__global__ void pool_accum(const float* __restrict__ H, const int* __restrict__ batch,
                           float* __restrict__ sums, float* __restrict__ cnts) {
    int node = (blockIdx.x * blockDim.x + threadIdx.x) >> 5;
    int lane = threadIdx.x & 31;
    if (node >= N_NODES) return;
    int gph = batch[node];
    const float4* hs = (const float4*)(H + (size_t)node * FDIM) + lane * 2;
    float4 v0 = hs[0], v1 = hs[1];
    float* s = sums + (size_t)gph * FDIM + lane * 8;
    atomicAdd(s + 0, v0.x); atomicAdd(s + 1, v0.y); atomicAdd(s + 2, v0.z); atomicAdd(s + 3, v0.w);
    atomicAdd(s + 4, v1.x); atomicAdd(s + 5, v1.y); atomicAdd(s + 6, v1.z); atomicAdd(s + 7, v1.w);
    if (lane == 0) atomicAdd(&cnts[gph], 1.0f);
}
__global__ void pool_div(const float* __restrict__ sums, const float* __restrict__ cnts,
                         float* __restrict__ emb) {
    int i = blockIdx.x * blockDim.x + threadIdx.x;
    if (i >= N_GRAPHS * FDIM) return;
    float c = fmaxf(cnts[i >> 8], 1.0f);
    emb[i] = sums[i] / c;
}

// ---------------- centroid head ----------------
__global__ void head(const float* __restrict__ emb, const float* __restrict__ cent,
                     float* __restrict__ out) {
    int gph  = blockIdx.x;           // 256 graphs
    int wave = threadIdx.x >> 5;
    int lane = threadIdx.x & 31;
    __shared__ float d2s[N_CLASSES * N_CENT];
    const float4* er = (const float4*)(emb + (size_t)gph * FDIM) + lane * 2;
    float4 e0 = er[0], e1 = er[1];
    for (int c = wave; c < N_CLASSES * N_CENT; c += 8) {
        const float4* cr = (const float4*)(cent + (size_t)c * FDIM) + lane * 2;
        float4 c0 = cr[0], c1 = cr[1];
        float p = 0.f, t;
        t = e0.x - c0.x; p += t * t;  t = e0.y - c0.y; p += t * t;
        t = e0.z - c0.z; p += t * t;  t = e0.w - c0.w; p += t * t;
        t = e1.x - c1.x; p += t * t;  t = e1.y - c1.y; p += t * t;
        t = e1.z - c1.z; p += t * t;  t = e1.w - c1.w; p += t * t;
        #pragma unroll
        for (int off = 16; off; off >>= 1) p += __shfl_xor(p, off, 32);
        if (lane == 0) d2s[c] = p;
    }
    __syncthreads();
    if (threadIdx.x < N_CLASSES) {
        int cls = threadIdx.x;
        float mn = fminf(d2s[cls * 3], fminf(d2s[cls * 3 + 1], d2s[cls * 3 + 2]));
        out[gph * N_CLASSES + cls] = -mn;
    }
}

// ---------------- host side ----------------
extern "C" void kernel_launch(void* const* d_in, const int* in_sizes, int n_in,
                              void* d_out, int out_size, void* d_ws, size_t ws_size,
                              hipStream_t stream) {
    (void)in_sizes; (void)n_in; (void)out_size; (void)ws_size;
    const float* x     = (const float*)d_in[0];
    const int*   ei    = (const int*)d_in[1];
    const int*   srcI  = ei;
    const int*   dstI  = ei + N_EDGES;
    const int*   batch = (const int*)d_in[2];
    const float* W[3]  = {(const float*)d_in[3], (const float*)d_in[5], (const float*)d_in[7]};
    const float* B[3]  = {(const float*)d_in[4], (const float*)d_in[6], (const float*)d_in[8]};
    const float* cent  = (const float*)d_in[9];
    const float* bnG[3] = {(const float*)d_in[10], (const float*)d_in[14], (const float*)d_in[18]};
    const float* bnB[3] = {(const float*)d_in[11], (const float*)d_in[15], (const float*)d_in[19]};
    const float* bnM[3] = {(const float*)d_in[12], (const float*)d_in[16], (const float*)d_in[20]};
    const float* bnV[3] = {(const float*)d_in[13], (const float*)d_in[17], (const float*)d_in[21]};

    // workspace layout (all offsets 256B-aligned)
    char* w8 = (char*)d_ws;
    size_t off = 0;
    float* bufA = (float*)(w8 + off); off += (size_t)N_NODES * FDIM * 4;        // 102.4 MB
    float* bufB = (float*)(w8 + off); off += (size_t)N_NODES * FDIM * 4;        // 102.4 MB
    float* dinv = (float*)(w8 + off); off += ((size_t)N_NODES * 4 + 255) & ~255ull;
    unsigned short* wpack = (unsigned short*)(w8 + off); off += (size_t)FDIM * FDIM * 2;
    float* sums = (float*)(w8 + off); off += (size_t)N_GRAPHS * FDIM * 4;
    float* cnts = (float*)(w8 + off); off += 256 * 4;
    float* emb  = (float*)(w8 + off); off += (size_t)N_GRAPHS * FDIM * 4;

    const int TB = 256;
    // degree / normalization
    deg_init <<<(N_NODES + TB - 1) / TB, TB, 0, stream>>>(dinv);
    deg_edges<<<(N_EDGES + TB - 1) / TB, TB, 0, stream>>>(dstI, dinv);
    deg_rsqrt<<<(N_NODES + TB - 1) / TB, TB, 0, stream>>>(dinv);

    const int elem4Blocks = (N_NODES * (FDIM / 4) + TB - 1) / TB;   // 25000
    for (int l = 0; l < 3; ++l) {
        const float* in = (l == 0) ? x : bufB;
        pack_w     <<<(FDIM * FDIM + TB - 1) / TB, TB, 0, stream>>>(W[l], wpack);
        gemm_bf16  <<<N_NODES / 16, TB, 0, stream>>>(in, wpack, bufA);
        self_loop  <<<elem4Blocks, TB, 0, stream>>>(bufA, dinv, bufB);
        edge_scatter<<<(N_EDGES * 32) / TB, TB, 0, stream>>>(bufA, bufB, srcI, dstI, dinv);
        bias_bn    <<<elem4Blocks, TB, 0, stream>>>(bufB, B[l], bnG[l], bnB[l], bnM[l], bnV[l],
                                                    (l < 2) ? 1 : 0);
    }

    pool_zero <<<(N_GRAPHS * FDIM + TB - 1) / TB, TB, 0, stream>>>(sums, cnts);
    pool_accum<<<(N_NODES * 32) / TB, TB, 0, stream>>>(bufB, batch, sums, cnts);
    pool_div  <<<(N_GRAPHS * FDIM + TB - 1) / TB, TB, 0, stream>>>(sums, cnts, emb);
    head      <<<N_GRAPHS, TB, 0, stream>>>(emb, cent, (float*)d_out);
}
